// FFT_Conv_Layer_19859928777106
// MI455X (gfx1250) — matrix operand order repaired and checked by reading, not compiled, not run on hardware
//
#include <hip/hip_runtime.h>

// CDNA5 / gfx1250: wave32, WMMA 16x16x32 f16 -> f32 accum.
typedef __attribute__((ext_vector_type(16))) _Float16 v16h;
typedef __attribute__((ext_vector_type(8)))  _Float16 v8h;
typedef __attribute__((ext_vector_type(8)))  float    v8f;
typedef __attribute__((ext_vector_type(4)))  float    v4f;

#define B_     16
#define C_     64   // inC == outC == 64
#define HW_    64
#define TILE   16   // 16 rows x 16 cols per block
#define HALO   18   // TILE + 2 (3x3 halo)
#define NQ     18   // 9 taps x 2 k-chunks

// sched_group_barrier masks (LLVM AMDGPU): 0x8 = MFMA/WMMA, 0x100 = DS read,
// 0x20 = VMEM read.
#if __has_builtin(__builtin_amdgcn_sched_group_barrier)
#define SGB(mask, size) __builtin_amdgcn_sched_group_barrier((mask), (size), 0)
#else
#define SGB(mask, size)
#endif

// out[b,o,y,x] = sum_{i, dy,dx in [-1,1]} img[b,i,y+dy,x+dx] * filt_re[i,o,1-dy,1-dx]
// (full linear conv + crop == "same" conv with flipped taps; imag plane is dead.)
__global__ __launch_bounds__(128)
void fft_conv3x3_wmma(const float* __restrict__ imgs,
                      const float* __restrict__ filts,   // [1,64,64,3,3,2] flat
                      float* __restrict__ out) {
  // f16 image tile, channel-fastest so A-fragments are two ds_load_b128 each.
  __shared__ __align__(16) _Float16 ilds[HALO * HALO * C_];

  const int b   = blockIdx.z;
  const int y0  = blockIdx.y * TILE;
  const int x0  = blockIdx.x * TILE;
  const int tid  = threadIdx.x;
  const int lane = tid & 31;
  const int wid  = tid >> 5;           // 4 waves: each owns 16 outC

  // Hint weights into cache early (global_prefetch_b8). Weights = 73728 floats.
  __builtin_prefetch(filts + tid * 288, 0, 1);

  // ---- Stage 18x18x64 halo tile into LDS as f16 (x fastest in global reads) ----
  const float* ib = imgs + (size_t)b * C_ * HW_ * HW_;
  for (int idx = tid; idx < HALO * HALO * C_; idx += 128) {
    const int c   = idx / (HALO * HALO);
    const int rem = idx - c * (HALO * HALO);
    const int rr  = rem / HALO;
    const int p   = rem - rr * HALO;
    const int gy  = y0 - 1 + rr;
    const int gx  = x0 - 1 + p;
    float v = 0.0f;
    if (gy >= 0 && gy < HW_ && gx >= 0 && gx < HW_)
      v = ib[(c * HW_ + gy) * HW_ + gx];
    ilds[(rr * HALO + p) * C_ + c] = (_Float16)v;
  }
  __syncthreads();

  // ---- WMMA main loop: D(16px x 16oc) += A(16px x 32ic) * B(32ic x 16oc) ----
  const int half = lane >> 4;          // 0: lanes 0-15, 1: lanes 16-31
  const int l    = lane & 15;
  const int ogb  = wid * 16;           // outC base for this wave
  const float* wrow = filts + (size_t)(ogb + l) * 18;  // o-row of weights

  v8f zero = {};
  v8f acc[TILE];
  #pragma unroll
  for (int i = 0; i < TILE; ++i) acc[i] = zero;

  // iteration q = (tap, kc): tap = q>>1, kc = (q&1)*32
  // B element addr (per lane): wrow[(2-tr)*6 + (2-tc)*2 + (kc + 8*half + (j&7) + 16*(j>>3))*1152]
  float wnxt[16];  // staged f32 weights for iteration q (loads issued at q-1)
  {
    #pragma unroll
    for (int j = 0; j < 16; ++j)       // prologue: loads for q = 0 (tap 0, kc 0)
      wnxt[j] = wrow[2 * 6 + 2 * 2 + (size_t)(half * 8 + (j & 7) + (j >> 3) * 16) * 1152];
  }

  #pragma unroll 1
  for (int q = 0; q < NQ; ++q) {
    const int tap = q >> 1;
    const int kc  = (q & 1) * 32;
    const int tr  = tap / 3;
    const int tc  = tap - tr * 3;

    // ---- consume staged weights -> B fragment (waits on last iter's loads) ----
    v16h bf;
    #pragma unroll
    for (int j = 0; j < 16; ++j) bf[j] = (_Float16)wnxt[j];

    // ---- issue global weight loads for q+1 (clamped: no branch) ----
    {
      const int qn  = (q < NQ - 1) ? q + 1 : q;
      const int tpn = qn >> 1;
      const int kcn = (qn & 1) * 32;
      const int trn = tpn / 3;
      const int tcn = tpn - trn * 3;
      const float* wp = wrow + (2 - trn) * 6 + (2 - tcn) * 2;
      #pragma unroll
      for (int j = 0; j < 16; ++j)
        wnxt[j] = wp[(size_t)(kcn + half * 8 + (j & 7) + (j >> 3) * 16) * 1152];
    }

    // ---- A rotation: 4 slots, prefetch distance 2 ----
    const _Float16* abase =
        &ilds[(size_t)(tr * HALO + (l + tc)) * C_ + kc + half * 8];
    v8h alo[4], ahi[4];
    #pragma unroll
    for (int s = 0; s < 2; ++s) {                    // prologue: frags 0,1
      const _Float16* ap = abase + (size_t)s * (HALO * C_);
      alo[s] = *(const v8h*)ap;                      // ds_load_b128
      ahi[s] = *(const v8h*)(ap + 16);               // ds_load_b128
    }
    #pragma unroll
    for (int mt = 0; mt < TILE; ++mt) {
      if (mt + 2 < TILE) {                           // compile-time under unroll
        const _Float16* ap = abase + (size_t)(mt + 2) * (HALO * C_);
        alo[(mt + 2) & 3] = *(const v8h*)ap;
        ahi[(mt + 2) & 3] = *(const v8h*)(ap + 16);
      }
      v16h af = __builtin_shufflevector(alo[mt & 3], ahi[mt & 3],
          0, 1, 2, 3, 4, 5, 6, 7, 8, 9, 10, 11, 12, 13, 14, 15);
      acc[mt] = __builtin_amdgcn_wmma_f32_16x16x32_f16(
          /*neg_a=*/false, af, /*neg_b=*/false, bf,
          /*c_mod=*/(short)0, acc[mt], /*reuse_a=*/false, /*reuse_b=*/false);
    }

    // ---- Enforce the distance-2 DS/WMMA pipeline on this basic block ----
    // 4 DS reads (frags 0,1) ahead of the first WMMA; then each WMMA overlaps
    // the DS pair for frag mt+2 and one global weight load for the next tap.
    SGB(0x100, 4);                       // DS read x4 (prologue frags)
    #pragma unroll
    for (int s = 0; s < 14; ++s) {
      SGB(0x008, 1);                     // WMMA x1
      SGB(0x100, 2);                     // DS read x2 (frag s+2)
      SGB(0x020, 1);                     // VMEM read x1 (next-tap weight)
    }
    SGB(0x008, 2);                       // final WMMA x2
    SGB(0x020, 2);                       // remaining weight loads
  }

  // ---- Store: D VGPR g, lane(half,l) -> (M = g + 8*half, N = l).
  // A lane's 8 acc elements are 8 consecutive x positions -> 2x b128 stores.
  float* ob = out + ((size_t)(b * C_ + ogb + l) * HW_ + y0) * HW_ + x0 + half * 8;
  #pragma unroll
  for (int mt = 0; mt < TILE; ++mt) {
    v4f s0 = {acc[mt][0], acc[mt][1], acc[mt][2], acc[mt][3]};
    v4f s1 = {acc[mt][4], acc[mt][5], acc[mt][6], acc[mt][7]};
    *(v4f*)(ob + (size_t)mt * HW_)     = s0;         // global_store_b128
    *(v4f*)(ob + (size_t)mt * HW_ + 4) = s1;         // global_store_b128
  }
}

extern "C" void kernel_launch(void* const* d_in, const int* in_sizes, int n_in,
                              void* d_out, int out_size, void* d_ws, size_t ws_size,
                              hipStream_t stream) {
  const float* imgs  = (const float*)d_in[0];   // [16,64,64,64] f32
  const float* filts = (const float*)d_in[1];   // [1,64,64,3,3,2] f32
  float* out = (float*)d_out;                   // [16,64,64,64] f32

  dim3 grid(HW_ / TILE, HW_ / TILE, B_);        // 4 x 4 x 16
  dim3 block(128);                              // 4 wave32
  fft_conv3x3_wmma<<<grid, block, 0, stream>>>(imgs, filts, out);
}
